// CausalMultiheadSelfAttention_19731079758669
// MI455X (gfx1250) — compile-verified, matrix operand order checked by reading, not run on hardware
//
#include <hip/hip_runtime.h>

typedef __attribute__((ext_vector_type(16))) _Float16 v16h;
typedef __attribute__((ext_vector_type(8)))  float    v8f;
typedef int v4i32 __attribute__((vector_size(16)));   // matches builtin param type

#define L_SEQ   2048
#define D_MODEL 1024
#define N_HEADS 16
#define D_HEAD  64
#define N_BATCH 2

// ---------------------------------------------------------------------------
// Async global->LDS staging (ASYNCcnt path), probe-confirmed signature:
//   (int4 AS(1)* src, int4 AS(3)* dst, imm offset, imm cpol)
// ---------------------------------------------------------------------------
#if __has_builtin(__builtin_amdgcn_global_load_async_to_lds_b128) && \
    __has_builtin(__builtin_amdgcn_s_wait_asynccnt)
#define HAS_ASYNC_LDS 1
#endif

static __device__ __forceinline__ void copy16B(const _Float16* g, _Float16* l) {
#if defined(HAS_ASYNC_LDS)
  __builtin_amdgcn_global_load_async_to_lds_b128(
      (__attribute__((address_space(1))) v4i32*)(uintptr_t)g,
      (__attribute__((address_space(3))) v4i32*)(uintptr_t)l,
      0, 0);
#else
  *(uint4*)l = *(const uint4*)g;
#endif
}
static __device__ __forceinline__ void staging_fence() {
#if defined(HAS_ASYNC_LDS)
  __builtin_amdgcn_s_wait_asynccnt(0);   // wait own wave's async loads -> LDS
#endif
}

// ---------------------------------------------------------------------------
// Fragment load, CDNA5 16-bit A-matrix 16x32 layout (ISA 7.12.2):
//   lanes 0-15 : row = lane,     K = [0..8) U [16..24)
//   lanes 16-31: row = lane-16,  K = [8..16) U [24..32)
// Dense 16-bit B (32x16) uses the symmetric layout (lane = N column, same K
// striping), so one loader serves both sides for row-major operands.
// ---------------------------------------------------------------------------
static __device__ __forceinline__ v16h load_frag16(const _Float16* rowptr, int lane) {
  const int kofs = (lane & 16) ? 8 : 0;
  union { uint4 u[2]; v16h v; } t;
  t.u[0] = *(const uint4*)(rowptr + kofs);
  t.u[1] = *(const uint4*)(rowptr + 16 + kofs);
  return t.v;
}

static __device__ __forceinline__ v8f wmma16(v16h a, v16h b, v8f c) {
  return __builtin_amdgcn_wmma_f32_16x16x32_f16(false, a, false, b, (short)0, c,
                                                false, false);
}

// reduce within each 16-lane half (xor masks 1,2,4,8) — matches wave32
// C-matrix row ownership (row = vgpr + 8*(lane>=16), col = lane&15).
static __device__ __forceinline__ float hmax16(float v) {
  v = fmaxf(v, __shfl_xor(v, 1, 32));
  v = fmaxf(v, __shfl_xor(v, 2, 32));
  v = fmaxf(v, __shfl_xor(v, 4, 32));
  v = fmaxf(v, __shfl_xor(v, 8, 32));
  return v;
}
static __device__ __forceinline__ float hsum16(float v) {
  v += __shfl_xor(v, 1, 32);
  v += __shfl_xor(v, 2, 32);
  v += __shfl_xor(v, 4, 32);
  v += __shfl_xor(v, 8, 32);
  return v;
}

// ---------------------------------------------------------------------------
__global__ void cvt_f32_to_f16_kernel(const float* __restrict__ src,
                                      _Float16* __restrict__ dst, int n) {
  int i = blockIdx.x * blockDim.x + threadIdx.x;
  if (i < n) dst[i] = (_Float16)src[i];
}

// ---------------------------------------------------------------------------
// Projection GEMM: out[m][n] = sum_k A[m][k] * W[n][k]   (M=4096, K=N=1024)
// 256 threads = 8 waves; block tile 128x64; wave tile 16x64.
// W tile (64 x 128-K chunk) async-staged in LDS once per block (8x reuse);
// A fragments double-buffered across K chunks (global latency overlap);
// B fragments double-buffered across ks steps (LDS latency overlap).
// mode 0: f16 out [b][h][l][dd] (Q,K)   mode 1: f16 out [b][h][dd][l] (V^T)
// mode 2: f32 out row-major (final projection)
// ---------------------------------------------------------------------------
__global__ void __launch_bounds__(256)
proj_gemm_kernel(const _Float16* __restrict__ A, const _Float16* __restrict__ W,
                 _Float16* __restrict__ out16, float* __restrict__ out32, int mode) {
  __shared__ __align__(16) _Float16 sW[64][136];   // 64 rows x 128 K, padded

  const int wave = threadIdx.x >> 5;
  const int lane = threadIdx.x & 31;
  const int colc = lane & 15;
  const int hlf  = lane >> 4;
  const int m0 = blockIdx.x * 128 + wave * 16;
  const int n0 = blockIdx.y * 64;

  const _Float16* Arow = A + (size_t)(m0 + colc) * D_MODEL;

  v8f acc[4] = {};
  v16h a[4];
#pragma unroll
  for (int s = 0; s < 4; ++s) a[s] = load_frag16(Arow + s * 32, lane);

  for (int kc = 0; kc < D_MODEL; kc += 128) {
    __syncthreads();                               // sW reuse guard
    // stage W[n0..n0+63][kc..kc+127]: 1024 x 16B, 4 per thread (ASYNCcnt)
    for (int i = threadIdx.x; i < 1024; i += 256) {
      const int row = i >> 4, c = i & 15;
      copy16B(W + (size_t)(n0 + row) * D_MODEL + kc + c * 8, &sW[row][c * 8]);
    }
    // prefetch next chunk's A fragments into registers (overlaps staging+MMA)
    v16h an[4];
    if (kc + 128 < D_MODEL) {
#pragma unroll
      for (int s = 0; s < 4; ++s)
        an[s] = load_frag16(Arow + kc + 128 + s * 32, lane);
    } else {
#pragma unroll
      for (int s = 0; s < 4; ++s) an[s] = a[s];
    }
    staging_fence();
    __syncthreads();

    // B-fragment double buffer: ks+1 ds loads in flight during ks WMMAs
    v16h bc[4], bn[4];
#pragma unroll
    for (int j = 0; j < 4; ++j)
      bc[j] = load_frag16((const _Float16*)&sW[j * 16 + colc][0], lane);
#pragma unroll
    for (int ks = 0; ks < 4; ++ks) {
      if (ks < 3) {
#pragma unroll
        for (int j = 0; j < 4; ++j)
          bn[j] = load_frag16((const _Float16*)&sW[j * 16 + colc][(ks + 1) * 32], lane);
      }
#pragma unroll
      for (int j = 0; j < 4; ++j)
        acc[j] = wmma16(a[ks], bc[j], acc[j]);
      if (ks < 3) {
#pragma unroll
        for (int j = 0; j < 4; ++j) bc[j] = bn[j];
      }
    }
#pragma unroll
    for (int s = 0; s < 4; ++s) a[s] = an[s];
  }

#pragma unroll
  for (int j = 0; j < 4; ++j) {
#pragma unroll
    for (int r = 0; r < 8; ++r) {
      const int m = m0 + r + 8 * hlf;          // token index (b*L + l)
      const int n = n0 + j * 16 + colc;        // feature index (h*64 + dd)
      const float val = acc[j][r];
      if (mode == 2) {
        out32[(size_t)m * D_MODEL + n] = val;
      } else {
        const int b = m >> 11, l = m & (L_SEQ - 1);
        const int hh = n >> 6, dd = n & 63;
        if (mode == 0)
          out16[((size_t)(b * N_HEADS + hh) * L_SEQ + l) * D_HEAD + dd] = (_Float16)val;
        else
          out16[((size_t)(b * N_HEADS + hh) * D_HEAD + dd) * L_SEQ + l] = (_Float16)val;
      }
    }
  }
}

// ---------------------------------------------------------------------------
// RoPE on [b][h][l][64] f16 tensor, one thread per pair.
// ---------------------------------------------------------------------------
__global__ void rope_kernel(_Float16* __restrict__ T, const int* __restrict__ pos) {
  const int total = N_BATCH * N_HEADS * L_SEQ * (D_HEAD / 2);
  int idx = blockIdx.x * blockDim.x + threadIdx.x;
  if (idx >= total) return;
  const int i  = idx & 31;
  const int l  = (idx >> 5) & (L_SEQ - 1);
  const int bh = idx >> 16;                      // 32*2048 = 2^16
  const float p = (float)pos[l];
  const float freq = __expf(-(float)i * (9.210340371976184f / 32.0f)); // 1e4^(-2i/64)
  float s, c;
  __sincosf(p * freq, &s, &c);
  const size_t base = ((size_t)bh * L_SEQ + l) * D_HEAD + 2 * i;
  const float x1 = (float)T[base];
  const float x2 = (float)T[base + 1];
  T[base]     = (_Float16)(x1 * c - x2 * s);
  T[base + 1] = (_Float16)(x1 * s + x2 * c);
}

// ---------------------------------------------------------------------------
// Causal flash attention. Block = (b, h, 128-query tile), 8 waves; each wave
// owns 16 query rows. K/V tiles (128 keys) async-staged in LDS, 32-key steps.
// ---------------------------------------------------------------------------
__global__ void __launch_bounds__(256)
flash_attn_kernel(const _Float16* __restrict__ Q, const _Float16* __restrict__ K,
                  const _Float16* __restrict__ Vt, _Float16* __restrict__ attn_out) {
  __shared__ __align__(16) _Float16 sK[128][72];
  __shared__ __align__(16) _Float16 sV[64][136];
  __shared__ __align__(16) _Float16 sP[8][16][32];

  const int qt = blockIdx.x, h = blockIdx.y, b = blockIdx.z;
  const int wave = threadIdx.x >> 5;
  const int lane = threadIdx.x & 31;
  const int colc = lane & 15;
  const int hlf  = lane >> 4;
  const int q0 = qt * 128 + wave * 16;

  const _Float16* Qbh = Q  + (size_t)(b * N_HEADS + h) * L_SEQ * D_HEAD;
  const _Float16* Kbh = K  + (size_t)(b * N_HEADS + h) * L_SEQ * D_HEAD;
  const _Float16* Vbh = Vt + (size_t)(b * N_HEADS + h) * D_HEAD * L_SEQ;

  const v16h aq0 = load_frag16(Qbh + (size_t)(q0 + colc) * D_HEAD,      lane);
  const v16h aq1 = load_frag16(Qbh + (size_t)(q0 + colc) * D_HEAD + 32, lane);

  v8f o0 = {}, o1 = {}, o2 = {}, o3 = {};
  float mrow[8], lsum[8];
#pragma unroll
  for (int r = 0; r < 8; ++r) { mrow[r] = -__builtin_inff(); lsum[r] = 0.0f; }

  for (int kt = 0; kt <= qt; ++kt) {             // uniform trip count / block
    const int kb0 = kt * 128;
    __syncthreads();
    for (int i = threadIdx.x; i < 1024; i += 256) {        // K tile 16 KB
      const int row = i >> 3, c = i & 7;
      copy16B(Kbh + (size_t)(kb0 + row) * D_HEAD + c * 8, &sK[row][c * 8]);
    }
    for (int i = threadIdx.x; i < 1024; i += 256) {        // V tile 16 KB
      const int row = i >> 4, c = i & 15;
      copy16B(Vbh + (size_t)row * L_SEQ + kb0 + c * 8, &sV[row][c * 8]);
    }
    if (kt < qt)
      __builtin_prefetch(Kbh + (size_t)(kb0 + 128) * D_HEAD, 0, 1);
    staging_fence();
    __syncthreads();

    for (int ks = 0; ks < 4; ++ks) {
      const int kb = ks * 32;
      // preload all S-side K fragments, then run the WMMA chain
      const v16h bk0 = load_frag16(&sK[kb + colc][0],       lane);
      const v16h bk1 = load_frag16(&sK[kb + colc][32],      lane);
      const v16h bk2 = load_frag16(&sK[kb + 16 + colc][0],  lane);
      const v16h bk3 = load_frag16(&sK[kb + 16 + colc][32], lane);
      v8f s0 = {}, s1 = {};
      s0 = wmma16(aq0, bk0, s0);
      s0 = wmma16(aq1, bk1, s0);
      s1 = wmma16(aq0, bk2, s1);
      s1 = wmma16(aq1, bk3, s1);

      // V fragments are independent of sP: issue now, complete during softmax
      const v16h bv0 = load_frag16(&sV[ 0 + colc][kb], lane);
      const v16h bv1 = load_frag16(&sV[16 + colc][kb], lane);
      const v16h bv2 = load_frag16(&sV[32 + colc][kb], lane);
      const v16h bv3 = load_frag16(&sV[48 + colc][kb], lane);

      float p0[8], p1[8];
      const int key0 = kb0 + kb + colc;
      const int key1 = key0 + 16;
#pragma unroll
      for (int r = 0; r < 8; ++r) {
        const int row = q0 + r + 8 * hlf;
        float v0 = s0[r] * 0.125f;               // 1/sqrt(64)
        float v1 = s1[r] * 0.125f;
        if (key0 > row) v0 = -__builtin_inff();
        if (key1 > row) v1 = -__builtin_inff();
        p0[r] = v0; p1[r] = v1;
      }
#pragma unroll
      for (int r = 0; r < 8; ++r) {
        const float mnew = fmaxf(mrow[r], hmax16(fmaxf(p0[r], p1[r])));
        const float corr = __expf(mrow[r] - mnew);
        p0[r] = __expf(p0[r] - mnew);
        p1[r] = __expf(p1[r] - mnew);
        const float rs = hsum16(p0[r] + p1[r]);
        lsum[r] = lsum[r] * corr + rs;
        mrow[r] = mnew;
        o0[r] *= corr; o1[r] *= corr; o2[r] *= corr; o3[r] *= corr;
      }
      // C-layout -> A-layout via per-wave LDS (same-wave DS ops are in-order)
      _Float16* pt = &sP[wave][0][0];
#pragma unroll
      for (int r = 0; r < 8; ++r) {
        pt[(r + 8 * hlf) * 32 + colc]      = (_Float16)p0[r];
        pt[(r + 8 * hlf) * 32 + 16 + colc] = (_Float16)p1[r];
      }
      const v16h pa = load_frag16(pt + colc * 32, lane);
      o0 = wmma16(pa, bv0, o0);
      o1 = wmma16(pa, bv1, o1);
      o2 = wmma16(pa, bv2, o2);
      o3 = wmma16(pa, bv3, o3);
    }
  }

#pragma unroll
  for (int r = 0; r < 8; ++r) {
    const int row = q0 + r + 8 * hlf;
    const float inv = 1.0f / lsum[r];
    const size_t base = ((size_t)b * L_SEQ + row) * D_MODEL + h * D_HEAD;
    attn_out[base +  0 + colc] = (_Float16)(o0[r] * inv);
    attn_out[base + 16 + colc] = (_Float16)(o1[r] * inv);
    attn_out[base + 32 + colc] = (_Float16)(o2[r] * inv);
    attn_out[base + 48 + colc] = (_Float16)(o3[r] * inv);
  }
}

// ---------------------------------------------------------------------------
extern "C" void kernel_launch(void* const* d_in, const int* in_sizes, int n_in,
                              void* d_out, int out_size, void* d_ws, size_t ws_size,
                              hipStream_t stream) {
  (void)in_sizes; (void)n_in; (void)out_size; (void)ws_size;
  const float* x  = (const float*)d_in[0];
  const float* Wq = (const float*)d_in[1];
  const float* Wk = (const float*)d_in[2];
  const float* Wv = (const float*)d_in[3];
  const float* Wo = (const float*)d_in[4];
  const int*  pos = (const int*)d_in[5];
  float* out = (float*)d_out;

  char* ws = (char*)d_ws;
  const size_t MB = 1024 * 1024;
  _Float16* xb   = (_Float16*)(ws + 0 * MB);
  _Float16* wqh  = (_Float16*)(ws + 8 * MB);
  _Float16* wkh  = (_Float16*)(ws + 10 * MB);
  _Float16* wvh  = (_Float16*)(ws + 12 * MB);
  _Float16* woh  = (_Float16*)(ws + 14 * MB);
  _Float16* Qb   = (_Float16*)(ws + 16 * MB);
  _Float16* Kb   = (_Float16*)(ws + 24 * MB);
  _Float16* Vtb  = (_Float16*)(ws + 32 * MB);
  _Float16* attn = (_Float16*)(ws + 40 * MB);

  const int BL = N_BATCH * L_SEQ;
  const int nX = BL * D_MODEL;
  const int nW = D_MODEL * D_MODEL;
  cvt_f32_to_f16_kernel<<<(nX + 255) / 256, 256, 0, stream>>>(x,  xb,  nX);
  cvt_f32_to_f16_kernel<<<(nW + 255) / 256, 256, 0, stream>>>(Wq, wqh, nW);
  cvt_f32_to_f16_kernel<<<(nW + 255) / 256, 256, 0, stream>>>(Wk, wkh, nW);
  cvt_f32_to_f16_kernel<<<(nW + 255) / 256, 256, 0, stream>>>(Wv, wvh, nW);
  cvt_f32_to_f16_kernel<<<(nW + 255) / 256, 256, 0, stream>>>(Wo, woh, nW);

  dim3 gproj(BL / 128, D_MODEL / 64);
  proj_gemm_kernel<<<gproj, 256, 0, stream>>>(xb, wqh, Qb,  nullptr, 0);
  proj_gemm_kernel<<<gproj, 256, 0, stream>>>(xb, wkh, Kb,  nullptr, 0);
  proj_gemm_kernel<<<gproj, 256, 0, stream>>>(xb, wvh, Vtb, nullptr, 1);

  const int npairs = N_BATCH * N_HEADS * L_SEQ * (D_HEAD / 2);
  rope_kernel<<<(npairs + 255) / 256, 256, 0, stream>>>(Qb, pos);
  rope_kernel<<<(npairs + 255) / 256, 256, 0, stream>>>(Kb, pos);

  flash_attn_kernel<<<dim3(L_SEQ / 128, N_HEADS, N_BATCH), 256, 0, stream>>>(
      Qb, Kb, Vtb, attn);

  proj_gemm_kernel<<<gproj, 256, 0, stream>>>(attn, woh, nullptr, out, 2);
}
